// Decoder_41815801593945
// MI455X (gfx1250) — matrix operand order repaired and checked by reading, not compiled, not run on hardware
//
#include <hip/hip_runtime.h>
#include <hip/hip_bf16.h>
#include <math.h>

// ---------------------------------------------------------------------------
// MI455X (gfx1250) fused MLP decoder, round 3.
// Compute bound (~480 GFLOP vs ~0.5 GB) -> bf16 WMMA with f32 accumulate.
// MT=32 rows/block (each B tile feeds 2 WMMAs); B tiles software-pipelined
// TWO tiles ahead so each s_wait_loadcnt tolerates a full L2 round trip
// behind 4 outstanding b128 loads.
// ---------------------------------------------------------------------------

typedef __attribute__((ext_vector_type(16))) __bf16 v16bf;
typedef __attribute__((ext_vector_type(8)))  __bf16 v8bf;
typedef __attribute__((ext_vector_type(8)))  float  v8f;

#define N_NODES 50000
#define N_EDGES 100000
#define LATENT  512
#define HID     1024
#define RBF_DIM 32
#define RBF_DMAXF 10.0f
#define BONDK   (LATENT + RBF_DIM)   // 544
#define HPAD    (HID + 8)            // LDS row stride (conflict-free)
#define MT      32                   // rows per block

#define WMMA_BF16(a, b, c) \
  __builtin_amdgcn_wmma_f32_16x16x32_bf16(false, (a), false, (b), (short)0, (c), false, false)

// ----------------------- weight prep: fp32 [K][N] -> bf16 [Npad][K] --------
__global__ __launch_bounds__(256) void transpose_bf16(
    const float* __restrict__ w, __bf16* __restrict__ wt,
    int K, int N, int Npad) {
  int idx = blockIdx.x * 256 + threadIdx.x;
  if (idx >= Npad * K) return;
  int n = idx / K, k = idx - n * K;
  float v = (n < N) ? w[(size_t)k * N + n] : 0.0f;
  wt[idx] = (__bf16)v;
}

// ------------------------------ tile loads ---------------------------------
__device__ __forceinline__ v16bf ld16(const __bf16* p, int hi_off) {
  union { v16bf v; v8bf h[2]; } u;
  u.h[0] = *(const v8bf*)p;
  u.h[1] = *(const v8bf*)(p + hi_off);
  return u.v;
}

__device__ __forceinline__ v8f vzero8() {
  return v8f{0.f,0.f,0.f,0.f,0.f,0.f,0.f,0.f};
}

// One hidden layer: 32 rows x 1024 cols; each of 8 waves owns a 128-col strip
// and two 16-row A tiles. B tiles register-pipelined 2 deep.
// Safe for Asrc == Hdst (barrier between last read and first store).
__device__ __forceinline__ void layer1024(
    const __bf16* __restrict__ Asrc, int astride,
    const __bf16* __restrict__ Wt, int kdim,
    const float* __restrict__ bias,
    __bf16* __restrict__ Hdst, int hstride,
    int wave, int lr, int lh) {
  v8f acc0[8], acc1[8];
#pragma unroll
  for (int t = 0; t < 8; ++t) { acc0[t] = vzero8(); acc1[t] = vzero8(); }

  const __bf16* arow0 = Asrc + lr * astride;          // rows 0..15
  const __bf16* arow1 = Asrc + (16 + lr) * astride;   // rows 16..31
  const int nsteps = kdim >> 5;
  const int nbase  = wave * 128 + lr;

  // flattened B tile index i -> col tile (i&7), k-step (i>>3)
  auto tileptr = [&](int i) {
    return Wt + (size_t)(nbase + (i & 7) * 16) * kdim + (i >> 3) * 32 + lh * 16;
  };

  // prologue: 2-deep B pipeline
  v16bf bp0 = ld16(tileptr(0), 8);
  v16bf bp1 = ld16(tileptr(1), 8);
  for (int ks = 0; ks < nsteps; ++ks) {
    v16bf a0 = ld16(arow0 + ks * 32 + lh * 16, 8);
    v16bf a1 = ld16(arow1 + ks * 32 + lh * 16, 8);
#pragma unroll
    for (int t = 0; t < 8; ++t) {
      // prefetch tile i+2 (tail prefetches are dead loads inside d_ws)
      v16bf bnext = ld16(tileptr(ks * 8 + t + 2), 8);
      acc0[t] = WMMA_BF16(a0, bp0, acc0[t]);
      acc1[t] = WMMA_BF16(a1, bp0, acc1[t]);
      bp0 = bp1;
      bp1 = bnext;
    }
  }
  __syncthreads();  // all waves done reading Asrc (may alias Hdst)
#pragma unroll
  for (int t = 0; t < 8; ++t) {
    int n = wave * 128 + t * 16 + lr;
    float bn = bias[n];
    __bf16* hcol = Hdst + n;
#pragma unroll
    for (int v = 0; v < 8; ++v) {
      float v0 = acc0[t][v] + bn;                 // rows  v+8*lh
      float v1 = acc1[t][v] + bn;                 // rows 16+v+8*lh
      v0 = v0 > 0.f ? v0 : 0.f;
      v1 = v1 > 0.f ? v1 : 0.f;
      hcol[(v + lh * 8) * hstride]        = (__bf16)v0;
      hcol[(16 + v + lh * 8) * hstride]   = (__bf16)v1;
    }
  }
}

// ------------------------------ fused MLP ----------------------------------
template <int K0, bool BOND>
__global__ __launch_bounds__(256) void mlp_fused(
    const float* __restrict__ xpos, const float* __restrict__ quant,
    const int* __restrict__ pair,
    const __bf16* __restrict__ w0t, const float* __restrict__ b0,
    const __bf16* __restrict__ w1t, const float* __restrict__ b1,
    const __bf16* __restrict__ w2t, const float* __restrict__ b2,
    float* __restrict__ out0, float* __restrict__ out1) {
  constexpr int XPAD = K0 + 8;
  __shared__ __attribute__((aligned(16))) __bf16 Xs[MT][XPAD];
  __shared__ __attribute__((aligned(16))) __bf16 Hs[MT][HPAD];
  __shared__ float Ds[MT * 16];
  __shared__ float dist_s[MT];
  __shared__ int   se[MT], de[MT];

  const int tid  = threadIdx.x;
  const int wave = tid >> 5;
  const int lane = tid & 31;
  const int lr   = lane & 15;
  const int lh   = lane >> 4;

  Ds[tid] = 0.f;
  Ds[tid + 256] = 0.f;

  // ---- build MT x K0 bf16 input tile in LDS ----
  if (BOND) {
    if (tid < MT) {
      int e = blockIdx.x * MT + tid;
      int s = pair[2 * e + 0], d = pair[2 * e + 1];
      se[tid] = s; de[tid] = d;
      float dx = xpos[3 * s + 0] - xpos[3 * d + 0];
      float dy = xpos[3 * s + 1] - xpos[3 * d + 1];
      float dz = xpos[3 * s + 2] - xpos[3 * d + 2];
      dist_s[tid] = sqrtf(dx * dx + dy * dy + dz * dz + 1e-8f);
    }
    __syncthreads();
#pragma unroll 1
    for (int r = 0; r < MT; ++r) {   // coalesced: 256 threads x float2 = 512 cols
      float2 a = ((const float2*)(quant + (size_t)se[r] * LATENT))[tid];
      float2 b = ((const float2*)(quant + (size_t)de[r] * LATENT))[tid];
      Xs[r][2 * tid + 0] = (__bf16)(a.x + b.x);
      Xs[r][2 * tid + 1] = (__bf16)(a.y + b.y);
    }
    for (int i = tid; i < MT * RBF_DIM; i += 256) {
      int e = i >> 5, j = i & 31;
      float mu = (RBF_DMAXF / 31.f) * (float)j;   // linspace(0,10,32)
      float sg = RBF_DMAXF / 32.f;
      float z  = (dist_s[e] - mu) / sg;
      Xs[e][LATENT + j] = (__bf16)expf(-z * z);
    }
  } else {
    int row0 = blockIdx.x * MT;
#pragma unroll 1
    for (int r = 0; r < MT; ++r) {
      int row = row0 + r;
      if (row < N_NODES) {
        float2 a = ((const float2*)(quant + (size_t)row * LATENT))[tid];
        Xs[r][2 * tid + 0] = (__bf16)a.x;
        Xs[r][2 * tid + 1] = (__bf16)a.y;
      } else {
        Xs[r][2 * tid + 0] = (__bf16)0.f;
        Xs[r][2 * tid + 1] = (__bf16)0.f;
      }
    }
  }
  __syncthreads();

  // ---- layer 1: [MT,K0] @ [K0,1024] -> Hs ----
  layer1024(&Xs[0][0], XPAD, w0t, K0, b0, &Hs[0][0], HPAD, wave, lr, lh);
  __syncthreads();
  // ---- layer 2: [MT,1024] @ [1024,1024] -> Hs (in place) ----
  layer1024(&Hs[0][0], HPAD, w1t, HID, b1, &Hs[0][0], HPAD, wave, lr, lh);
  __syncthreads();

  // ---- layer 3: [MT,1024] @ [1024,16], split-K across the 8 waves ----
  v8f acc0 = vzero8(), acc1 = vzero8();
  const __bf16* arow0 = &Hs[0][0] + lr * HPAD;
  const __bf16* arow1 = &Hs[0][0] + (16 + lr) * HPAD;
#pragma unroll
  for (int ks4 = 0; ks4 < 4; ++ks4) {
    int ks = wave * 4 + ks4;   // wave w reduces K in [w*128, w*128+128)
    v16bf b  = ld16(w2t + (size_t)lr * HID + ks * 32 + lh * 16, 8);
    v16bf a0 = ld16(arow0 + ks * 32 + lh * 16, 8);
    v16bf a1 = ld16(arow1 + ks * 32 + lh * 16, 8);
    acc0 = WMMA_BF16(a0, b, acc0);
    acc1 = WMMA_BF16(a1, b, acc1);
  }
#pragma unroll
  for (int v = 0; v < 8; ++v) {
    atomicAdd(&Ds[(v + lh * 8) * 16 + lr],      acc0[v]);   // ds_add_f32 reduce
    atomicAdd(&Ds[(16 + v + lh * 8) * 16 + lr], acc1[v]);
  }
  __syncthreads();

  // ---- epilogue: bias + scatter ----
  for (int i = tid; i < MT * 16; i += 256) {
    int m = i >> 4, n = i & 15;
    int row = blockIdx.x * MT + m;
    if (BOND) {
      if (n < 5) out0[(size_t)row * 5 + n] = Ds[i] + b2[n];
    } else if (row < N_NODES) {
      float val = Ds[i] + b2[n];
      if (n < 10) out0[(size_t)row * 10 + n] = val;
      else        out1[(size_t)row * 6 + (n - 10)] = val;
    }
  }
}

// ------------------------------ launcher -----------------------------------
extern "C" void kernel_launch(void* const* d_in, const int* in_sizes, int n_in,
                              void* d_out, int out_size, void* d_ws, size_t ws_size,
                              hipStream_t stream) {
  const float* xpos  = (const float*)d_in[0];
  const float* quant = (const float*)d_in[1];
  const int*   pair  = (const int*)  d_in[2];
  const float* aw0 = (const float*)d_in[3];  const float* ab0 = (const float*)d_in[4];
  const float* aw1 = (const float*)d_in[5];  const float* ab1 = (const float*)d_in[6];
  const float* aw2 = (const float*)d_in[7];  const float* ab2 = (const float*)d_in[8];
  const float* bw0 = (const float*)d_in[9];  const float* bb0 = (const float*)d_in[10];
  const float* bw1 = (const float*)d_in[11]; const float* bb1 = (const float*)d_in[12];
  const float* bw2 = (const float*)d_in[13]; const float* bb2 = (const float*)d_in[14];

  // bf16 transposed weights in workspace
  __bf16* ws    = (__bf16*)d_ws;
  __bf16* w0ta  = ws;                                // 1024 x 512
  __bf16* w1ta  = w0ta + (size_t)1024 * 512;         // 1024 x 1024
  __bf16* w2ta  = w1ta + (size_t)1024 * 1024;        //   16 x 1024
  __bf16* w0tb  = w2ta + (size_t)16   * 1024;        // 1024 x 544
  __bf16* w1tb  = w0tb + (size_t)1024 * BONDK;       // 1024 x 1024
  __bf16* w2tb  = w1tb + (size_t)1024 * 1024;        //   16 x 1024 (rows 5..15 zero)

  auto tr = [&](const float* w, __bf16* wt, int K, int N, int Npad) {
    int total = Npad * K;
    transpose_bf16<<<(total + 255) / 256, 256, 0, stream>>>(w, wt, K, N, Npad);
  };
  tr(aw0, w0ta, LATENT, HID, HID);
  tr(aw1, w1ta, HID, HID, HID);
  tr(aw2, w2ta, HID, 16, 16);
  tr(bw0, w0tb, BONDK, HID, HID);
  tr(bw1, w1tb, HID, HID, HID);
  tr(bw2, w2tb, HID, 5, 16);

  float* out        = (float*)d_out;
  float* atom_type  = out;                               // 50000 x 10
  float* atom_chg   = out + (size_t)N_NODES * 10;        // 50000 x 6
  float* bond_order = out + (size_t)N_NODES * 16;        // 100000 x 5

  mlp_fused<LATENT, false><<<(N_NODES + MT - 1) / MT, 256, 0, stream>>>(
      xpos, quant, pair, w0ta, ab0, w1ta, ab1, w2ta, ab2, atom_type, atom_chg);
  mlp_fused<BONDK, true><<<N_EDGES / MT, 256, 0, stream>>>(
      xpos, quant, pair, w0tb, bb0, w1tb, bb1, w2tb, bb2, bond_order, nullptr);
}
